// MultiHeadAttention_76038101008807
// MI455X (gfx1250) — matrix operand order repaired and checked by reading, not compiled, not run on hardware
//
#include <hip/hip_runtime.h>
#include <hip/hip_bf16.h>
#include <math.h>
#include <stdint.h>

// ---------------- problem constants ----------------
#define BB 2
#define SS 2048
#define FF 1024
#define HH 16
#define DD 64
#define LDA 1032   // padded LDS row stride in elements (2064 B: 16B-aligned, bank-conflict-free)
// ---------------------------------------------------

typedef __bf16 bf16_t;
typedef __attribute__((ext_vector_type(16))) __bf16 v16bf;
typedef __attribute__((ext_vector_type(8)))  float  v8f;
typedef __attribute__((ext_vector_type(4)))  unsigned int v4u;

union FragBF { v16bf v; v4u u[2]; };

#define WMMA_BF16(A, B, C) \
    __builtin_amdgcn_wmma_f32_16x16x32_bf16(false, (A), false, (B), (short)0, (C), false, false)

// A-fragment: 16x32 bf16 (M x K), row-major source with row stride `ld` elems.
// Layout: lane L holds row M=L&15; lanes 0-15: K=0..15,
// lanes 16-31: K=16..31 (matches load_b_frag). Two 16-byte loads per lane.
__device__ __forceinline__ v16bf load_a_frag(const bf16_t* base, int ld, int lane) {
    const int m  = lane & 15;
    const int k0 = (lane >> 4) * 16;
    const bf16_t* p = base + (size_t)m * ld + k0;
    FragBF f;
    f.u[0] = *(const v4u*)(p);
    f.u[1] = *(const v4u*)(p + 8);
    return f.v;
}

// B-fragment: 32x16 bf16 (K x N), sourced from a TRANSPOSED array bt[n][k]
// with row stride `ld`. Lane L holds col N=L&15; lanes 0-15 hold K=0..15,
// lanes 16-31 hold K=16..31 -> 16 contiguous elems per lane.
__device__ __forceinline__ v16bf load_b_frag(const bf16_t* base_t, int ld, int lane) {
    const int n  = lane & 15;
    const int k0 = (lane >> 4) * 16;
    const bf16_t* p = base_t + (size_t)n * ld + k0;
    FragBF f;
    f.u[0] = *(const v4u*)(p);
    f.u[1] = *(const v4u*)(p + 8);
    return f.v;
}

// Reductions across the 16-lane halves of the wave (C-layout rows live in a half).
__device__ __forceinline__ float half_max(float v) {
    v = fmaxf(v, __shfl_xor(v, 1, 32));
    v = fmaxf(v, __shfl_xor(v, 2, 32));
    v = fmaxf(v, __shfl_xor(v, 4, 32));
    v = fmaxf(v, __shfl_xor(v, 8, 32));
    return v;
}
__device__ __forceinline__ float half_sum(float v) {
    v += __shfl_xor(v, 1, 32);
    v += __shfl_xor(v, 2, 32);
    v += __shfl_xor(v, 4, 32);
    v += __shfl_xor(v, 8, 32);
    return v;
}

// Issue one 16-byte async global->LDS copy (per-lane, ASYNCcnt-tracked).
__device__ __forceinline__ void async_copy_b128(void* lds_dst, const void* gsrc) {
    unsigned ldst = (unsigned)(uintptr_t)lds_dst;   // wave-relative LDS byte offset
    asm volatile("global_load_async_to_lds_b128 %0, %1, off"
                 :: "v"(ldst), "v"(gsrc) : "memory");
}
__device__ __forceinline__ void wait_async_all() {
    asm volatile("s_wait_asynccnt 0" ::: "memory");
}

// ---------------------------------------------------------------------------
// Kernel 1: fp32 -> bf16 conversions. Weights are transposed so that WMMA
// B-fragments become contiguous per-lane loads.
// ---------------------------------------------------------------------------
__global__ void cvt_kernel(const float* __restrict__ x,
                           const float* __restrict__ wqkv,
                           const float* __restrict__ wout,
                           bf16_t* __restrict__ xb,
                           bf16_t* __restrict__ wqkvT,
                           bf16_t* __restrict__ woutT) {
    const size_t NX = (size_t)BB * SS * FF;
    const size_t NW = (size_t)FF * 3 * FF;
    const size_t NO = (size_t)FF * FF;
    const size_t total = NX + NW + NO;
    for (size_t i = (size_t)blockIdx.x * blockDim.x + threadIdx.x; i < total;
         i += (size_t)gridDim.x * blockDim.x) {
        if (i < NX) {
            xb[i] = (bf16_t)x[i];
        } else if (i < NX + NW) {
            size_t j = i - NX;                 // j = k*3F + n
            size_t k = j / (3 * FF), n = j % (3 * FF);
            wqkvT[n * FF + k] = (bf16_t)wqkv[j];
        } else {
            size_t j = i - NX - NW;            // j = k*F + n
            size_t k = j >> 10, n = j & (FF - 1);
            woutT[(n << 10) + k] = (bf16_t)wout[j];
        }
    }
}

// ---------------------------------------------------------------------------
// Kernel 2: QKV projection. C[4096 x 3072] = xb @ Wqkv + bias.
// Block = 8 waves sharing one 32-row A strip, async-staged into LDS once.
// Each wave: 32(M) x 64(N) tile = 8 accumulators; per K-chunk:
//   4 ds_load_b128 (A) + 8 global_load_b128 (B) feed 8 WMMAs.
// Epilogue scatters into Q[B,H,S,D], K[B,H,S,D], Vt[B,H,D,S] (bf16).
// ---------------------------------------------------------------------------
__global__ void __launch_bounds__(256)
qkv_gemm_kernel(const bf16_t* __restrict__ xb,
                const bf16_t* __restrict__ wt,   // [3F][F] (transposed)
                const float*  __restrict__ bias, // [3F]
                bf16_t* __restrict__ Qo,
                bf16_t* __restrict__ Ko,
                bf16_t* __restrict__ Vto) {
    const int lane = threadIdx.x & 31;
    const int wave = threadIdx.x >> 5;
    const int NBN  = (3 * FF) / 512;             // 6 N-blocks
    const int m0   = (blockIdx.x / NBN) * 32;
    const int n0   = (blockIdx.x % NBN) * 512 + wave * 64;

    __shared__ __align__(16) bf16_t lds_a[32][LDA];

    // ---- async-stage the 32 x 1024 A strip (64 KB) into LDS ----
    {
        const bf16_t* src = xb + (size_t)m0 * FF;
#pragma unroll
        for (int i = 0; i < 16; ++i) {
            const int c   = threadIdx.x + 256 * i;   // 0..4095 16B-chunks
            const int row = c >> 7;
            const int col = (c & 127) * 8;           // element column
            async_copy_b128(&lds_a[row][col], src + (size_t)row * FF + col);
        }
        wait_async_all();
        __syncthreads();
    }

    v8f acc[8];
#pragma unroll
    for (int t = 0; t < 8; ++t) acc[t] = 0.0f;

    for (int k = 0; k < FF; k += 32) {
        const v16bf a0 = load_a_frag(&lds_a[0][0]  + k, LDA, lane);
        const v16bf a1 = load_a_frag(&lds_a[16][0] + k, LDA, lane);
        if (k + 64 < FF)
            __builtin_prefetch(wt + (size_t)n0 * FF + k + 64, 0, 3);
#pragma unroll
        for (int t = 0; t < 4; ++t) {
            const v16bf bfr = load_b_frag(wt + (size_t)(n0 + t * 16) * FF + k, FF, lane);
            acc[t]     = WMMA_BF16(a0, bfr, acc[t]);
            acc[t + 4] = WMMA_BF16(a1, bfr, acc[t + 4]);
        }
    }

    const int hi = lane >> 4, nl = lane & 15;
#pragma unroll
    for (int t = 0; t < 4; ++t) {
        const int n    = n0 + t * 16 + nl;
        const float bv = bias[n];
        const int part = n >> 10;          // 0=Q 1=K 2=V
        const int f    = n & (FF - 1);
        const int h    = f >> 6;
        const int d    = f & (DD - 1);
#pragma unroll
        for (int ms = 0; ms < 2; ++ms) {
#pragma unroll
            for (int r = 0; r < 8; ++r) {
                const int m = m0 + ms * 16 + r + 8 * hi;
                const int b = m >> 11;     // / SS
                const int s = m & (SS - 1);
                bf16_t val = (bf16_t)(acc[t + 4 * ms][r] + bv);
                if (part == 0)
                    Qo[(((size_t)(b * HH + h)) * SS + s) * DD + d] = val;
                else if (part == 1)
                    Ko[(((size_t)(b * HH + h)) * SS + s) * DD + d] = val;
                else
                    Vto[(((size_t)(b * HH + h)) * DD + d) * SS + s] = val;
            }
        }
    }
}

// ---------------------------------------------------------------------------
// Kernel 3: flash attention with ALiBi + causal mask.
// One wave owns one 16-query tile of one (b,h); the 8 waves of a block are
// consecutive q-tiles of the SAME (b,h), so their K/V streams hit WGP$.
// QK^T: 4 WMMAs per 32-key chunk; P re-shaped through LDS; 4 PV WMMAs.
// ---------------------------------------------------------------------------
__global__ void __launch_bounds__(256)
attn_kernel(const bf16_t* __restrict__ Q,
            const bf16_t* __restrict__ Kb,
            const bf16_t* __restrict__ Vt,
            const float*  __restrict__ slopes,
            bf16_t* __restrict__ attnout) {
    const int lane = threadIdx.x & 31;
    const int wave = threadIdx.x >> 5;
    const int tile = blockIdx.x * 8 + wave;     // 0 .. B*H*(S/16)-1
    const int qt = tile & 127;                  // S/16 = 128
    const int bh = tile >> 7;
    const int h  = bh & (HH - 1);
    const int b  = bh >> 4;
    const int q0 = qt * 16;
    const int hi = lane >> 4, nl = lane & 15;

    __shared__ __align__(16) bf16_t lds_p[8][16][40];   // 80B row stride: conflict-free
    bf16_t (*P)[40] = lds_p[wave];

    const bf16_t* Qbh  = Q  + ((size_t)(b * HH + h)) * SS * DD;
    const bf16_t* Kbh  = Kb + ((size_t)(b * HH + h)) * SS * DD;
    const bf16_t* Vtbh = Vt + ((size_t)(b * HH + h)) * DD * SS;
    const float slope  = slopes[h];
    const float scale  = 0.125f;                // 1/sqrt(64)

    const v16bf qa0 = load_a_frag(Qbh + (size_t)q0 * DD +  0, DD, lane);
    const v16bf qa1 = load_a_frag(Qbh + (size_t)q0 * DD + 32, DD, lane);

    v8f o0 = 0.0f, o1 = 0.0f, o2 = 0.0f, o3 = 0.0f;
    float mrow[8], lrow[8];
#pragma unroll
    for (int r = 0; r < 8; ++r) { mrow[r] = -1e30f; lrow[r] = 0.0f; }

    const int kend = q0 + 16;                   // last key needed (exclusive)
    for (int kc = 0; kc < kend; kc += 32) {
        // ---- scores: 16 q x 32 keys ----
        v8f s0 = 0.0f, s1 = 0.0f;
        {
            v16bf kf;
            kf = load_b_frag(Kbh + (size_t)(kc)      * DD +  0, DD, lane);
            s0 = WMMA_BF16(qa0, kf, s0);
            kf = load_b_frag(Kbh + (size_t)(kc)      * DD + 32, DD, lane);
            s0 = WMMA_BF16(qa1, kf, s0);
            kf = load_b_frag(Kbh + (size_t)(kc + 16) * DD +  0, DD, lane);
            s1 = WMMA_BF16(qa0, kf, s1);
            kf = load_b_frag(Kbh + (size_t)(kc + 16) * DD + 32, DD, lane);
            s1 = WMMA_BF16(qa1, kf, s1);
        }

        // ---- ALiBi + causal + online softmax ----
        const int kv0 = kc + nl;
        const int kv1 = kc + 16 + nl;
#pragma unroll
        for (int r = 0; r < 8; ++r) {
            const int q = q0 + r + 8 * hi;
            float v0 = s0[r] * scale + slope * (float)(kv0 - q);
            float v1 = s1[r] * scale + slope * (float)(kv1 - q);
            if (kv0 > q) v0 = -1e30f;
            if (kv1 > q) v1 = -1e30f;
            const float mx = half_max(fmaxf(v0, v1));
            const float nm = fmaxf(mrow[r], mx);
            const float sc = __expf(mrow[r] - nm);
            const float p0 = __expf(v0 - nm);
            const float p1 = __expf(v1 - nm);
            lrow[r] = lrow[r] * sc + half_sum(p0 + p1);
            mrow[r] = nm;
            o0[r] *= sc; o1[r] *= sc; o2[r] *= sc; o3[r] *= sc;
            const int m = r + 8 * hi;
            P[m][nl]      = (bf16_t)p0;
            P[m][nl + 16] = (bf16_t)p1;
        }
        asm volatile("s_wait_dscnt 0" ::: "memory");

        // ---- P(16x32) @ V(32x64): 4 WMMAs ----
        const v16bf pa = load_a_frag(&P[0][0], 40, lane);
        v16bf vf;
        vf = load_b_frag(Vtbh + (size_t) 0 * SS + kc, SS, lane);
        o0 = WMMA_BF16(pa, vf, o0);
        vf = load_b_frag(Vtbh + (size_t)16 * SS + kc, SS, lane);
        o1 = WMMA_BF16(pa, vf, o1);
        vf = load_b_frag(Vtbh + (size_t)32 * SS + kc, SS, lane);
        o2 = WMMA_BF16(pa, vf, o2);
        vf = load_b_frag(Vtbh + (size_t)48 * SS + kc, SS, lane);
        o3 = WMMA_BF16(pa, vf, o3);
    }

    // ---- normalize + store attn output as bf16 [B*S, F] ----
    v8f outs[4] = {o0, o1, o2, o3};
#pragma unroll
    for (int r = 0; r < 8; ++r) {
        const float inv = 1.0f / lrow[r];
        const int srow = q0 + r + 8 * hi;
        bf16_t* dst = attnout + ((size_t)(b * SS + srow)) * FF + h * DD + nl;
#pragma unroll
        for (int t = 0; t < 4; ++t)
            dst[t * 16] = (bf16_t)(outs[t][r] * inv);
    }
}

// ---------------------------------------------------------------------------
// Kernel 4: output projection. out[4096 x 1024] = attn @ Wout + b_out (fp32).
// Same LDS-staged 32x64-per-wave structure as kernel 2.
// ---------------------------------------------------------------------------
__global__ void __launch_bounds__(256)
out_gemm_kernel(const bf16_t* __restrict__ ab,
                const bf16_t* __restrict__ wt,   // [F][F] transposed
                const float*  __restrict__ bias, // [F]
                float* __restrict__ out) {
    const int lane = threadIdx.x & 31;
    const int wave = threadIdx.x >> 5;
    const int NBN  = FF / 512;                  // 2 N-blocks
    const int m0   = (blockIdx.x / NBN) * 32;
    const int n0   = (blockIdx.x % NBN) * 512 + wave * 64;

    __shared__ __align__(16) bf16_t lds_a[32][LDA];

    {
        const bf16_t* src = ab + (size_t)m0 * FF;
#pragma unroll
        for (int i = 0; i < 16; ++i) {
            const int c   = threadIdx.x + 256 * i;
            const int row = c >> 7;
            const int col = (c & 127) * 8;
            async_copy_b128(&lds_a[row][col], src + (size_t)row * FF + col);
        }
        wait_async_all();
        __syncthreads();
    }

    v8f acc[8];
#pragma unroll
    for (int t = 0; t < 8; ++t) acc[t] = 0.0f;

    for (int k = 0; k < FF; k += 32) {
        const v16bf a0 = load_a_frag(&lds_a[0][0]  + k, LDA, lane);
        const v16bf a1 = load_a_frag(&lds_a[16][0] + k, LDA, lane);
        if (k + 64 < FF)
            __builtin_prefetch(wt + (size_t)n0 * FF + k + 64, 0, 3);
#pragma unroll
        for (int t = 0; t < 4; ++t) {
            const v16bf bfr = load_b_frag(wt + (size_t)(n0 + t * 16) * FF + k, FF, lane);
            acc[t]     = WMMA_BF16(a0, bfr, acc[t]);
            acc[t + 4] = WMMA_BF16(a1, bfr, acc[t + 4]);
        }
    }

    const int hi = lane >> 4, nl = lane & 15;
#pragma unroll
    for (int t = 0; t < 4; ++t) {
        const int n    = n0 + t * 16 + nl;
        const float bv = bias[n];
#pragma unroll
        for (int ms = 0; ms < 2; ++ms) {
#pragma unroll
            for (int r = 0; r < 8; ++r) {
                const int m = m0 + ms * 16 + r + 8 * hi;
                out[(size_t)m * FF + n] = acc[t + 4 * ms][r] + bv;
            }
        }
    }
}

// ---------------------------------------------------------------------------
extern "C" void kernel_launch(void* const* d_in, const int* in_sizes, int n_in,
                              void* d_out, int out_size, void* d_ws, size_t ws_size,
                              hipStream_t stream) {
    const float* x      = (const float*)d_in[0];
    const float* Wqkv   = (const float*)d_in[1];
    const float* bqkv   = (const float*)d_in[2];
    const float* Wout   = (const float*)d_in[3];
    const float* bout   = (const float*)d_in[4];
    const float* slopes = (const float*)d_in[5];
    float* out = (float*)d_out;

    // workspace carve-up (bytes)
    char* ws = (char*)d_ws;
    const size_t SZ_XB  = (size_t)BB * SS * FF * 2;        //  8 MB
    const size_t SZ_WQT = (size_t)3 * FF * FF * 2;         //  6 MB
    const size_t SZ_WOT = (size_t)FF * FF * 2;             //  2 MB
    const size_t SZ_QKV = (size_t)BB * HH * SS * DD * 2;   //  8 MB each
    bf16_t* xb     = (bf16_t*)(ws);                 ws += SZ_XB;
    bf16_t* wqkvT  = (bf16_t*)(ws);                 ws += SZ_WQT;
    bf16_t* woutT  = (bf16_t*)(ws);                 ws += SZ_WOT;
    bf16_t* Qb     = (bf16_t*)(ws);                 ws += SZ_QKV;
    bf16_t* Kb     = (bf16_t*)(ws);                 ws += SZ_QKV;
    bf16_t* Vtb    = (bf16_t*)(ws);                 ws += SZ_QKV;
    bf16_t* attnob = (bf16_t*)(ws);                 ws += SZ_XB;

    // 1) fp32 -> bf16 converts (+ weight transposes)
    cvt_kernel<<<1024, 256, 0, stream>>>(x, Wqkv, Wout, xb, wqkvT, woutT);

    // 2) QKV projection: (4096/32) x (3072/512) = 128 x 6 = 768 blocks
    qkv_gemm_kernel<<<768, 256, 0, stream>>>(xb, wqkvT, bqkv, Qb, Kb, Vtb);

    // 3) flash attention: B*H*(S/16) = 4096 q-tiles / 8 waves per block
    attn_kernel<<<512, 256, 0, stream>>>(Qb, Kb, Vtb, slopes, attnob);

    // 4) output projection: (4096/32) x (1024/512) = 128 x 2 = 256 blocks
    out_gemm_kernel<<<256, 256, 0, stream>>>(attnob, woutT, bout, out);
}
